// QMixtralDecoderLayer_58256936403178
// MI455X (gfx1250) — matrix operand order repaired and checked by reading, not compiled, not run on hardware
//
#include <hip/hip_runtime.h>
#include <cstdint>
#include <cstddef>

typedef __bf16 bf16;
typedef __attribute__((ext_vector_type(16))) __bf16 v16bf;
typedef __attribute__((ext_vector_type(8)))  float  v8f;
typedef __attribute__((ext_vector_type(4)))  unsigned u32x4;
typedef __attribute__((ext_vector_type(8)))  int      i32x8;
typedef __attribute__((ext_vector_type(4)))  int      i32x4;

static __device__ __forceinline__ v8f wmma_bf16(v16bf a, v16bf b, v8f c) {
  // D = A(16x32 bf16) * B(32x16 bf16) + C(16x16 f32)
  return __builtin_amdgcn_wmma_f32_16x16x32_bf16(false, a, false, b, (short)0, c,
                                                 false, false);
}

// ---------------------------------------------------------------------------
// Tensor Data Mover: 2D bf16 tile (tile_w elems x tile_h rows) from global to
// LDS, with per-row DWORD padding so LDS rows land on our padded strides.
// Descriptor bitfields per CDNA5 ISA.
//  pad_icode: DWORDs before pad = 2<<pad_icode ; pad_acode: pad = pad_acode+1 DW
// ---------------------------------------------------------------------------
static __device__ __forceinline__ void tdm_load_2d_bf16(
    unsigned lds_byte_addr, const void* gptr, unsigned tile_w, unsigned tile_h,
    unsigned long long row_stride_elems, unsigned tensor_w, unsigned tensor_h,
    unsigned pad_icode, unsigned pad_acode) {
  unsigned long long ga = (unsigned long long)(uintptr_t)gptr;
  u32x4 g0;
  g0[0] = 1u;                                    // count=1, user descriptor
  g0[1] = lds_byte_addr;                         // lds_addr [63:32]
  g0[2] = (unsigned)(ga & 0xffffffffull);        // global_addr low
  g0[3] = (unsigned)((ga >> 32) & 0x1ffffffull)  // global_addr high (57b)
          | (2u << 30);                          // type=2 ("image")
  i32x8 g1;
  unsigned w0 = (1u << 16)                       // data_size=1 -> 2 bytes
                | (1u << 20)                     // pad_enable
                | (pad_icode << 22) | (pad_acode << 25);
  g1[0] = (int)w0;                               // wg_mask=0, no atomic barrier
  g1[1] = (int)((tensor_w & 0xffffu) << 16);     // atomic_addr=0 | td0[15:0]
  g1[2] = (int)(((tensor_w >> 16) & 0xffffu) | ((tensor_h & 0xffffu) << 16));
  g1[3] = (int)(((tensor_h >> 16) & 0xffffu) | ((tile_w & 0xffffu) << 16));
  g1[4] = (int)(tile_h & 0xffffu);               // tile_dim1 | tile_dim2=0
  g1[5] = (int)(unsigned)(row_stride_elems & 0xffffffffull);   // td0_stride lo
  g1[6] = (int)(unsigned)((row_stride_elems >> 32) & 0xffffull);  // hi16
  g1[7] = 0;
  i32x4 gz = {0, 0, 0, 0};
#if __clang_major__ >= 23
  i32x8 gz8 = {0, 0, 0, 0, 0, 0, 0, 0};
  __builtin_amdgcn_tensor_load_to_lds(g0, g1, gz, gz, gz8, 0);
#else
  __builtin_amdgcn_tensor_load_to_lds(g0, g1, gz, gz, 0);
#endif
}

// ---------------------------------------------------------------------------
// Fragment loaders, contiguous k order shared by A and B (wave32, 16-bit operands)
// A 16x32: lanes 0-15 -> M=lane, K=0..15; lanes 16-31 -> K=16..31
// B 32x16: lane col = lane&15; lanes 0-15 -> K=0..15, lanes 16-31 -> K=16..31
// ---------------------------------------------------------------------------
static __device__ __forceinline__ v16bf ld_a(const bf16* base, int row0, int lane,
                                             int stride) {
  const unsigned* up = (const unsigned*)(base + (size_t)(row0 + (lane & 15)) * stride);
  const int kp = (lane >> 4) * 8;  // pair offset 0 or 8
  union { v16bf v; unsigned u[8]; } r;
#pragma unroll
  for (int j = 0; j < 8; ++j) r.u[j] = up[kp + j];
  return r.v;
}

static __device__ __forceinline__ v16bf ld_b(const bf16* base, int col0, int lane,
                                             int stride) {
  const unsigned* up = (const unsigned*)(base + (size_t)(col0 + (lane & 15)) * stride);
  const int kp = (lane < 16) ? 0 : 8;  // pair offset: K 0..15 or 16..31
  union { v16bf v; unsigned u[8]; } r;
#pragma unroll
  for (int j = 0; j < 8; ++j) r.u[j] = up[kp + j];
  return r.v;
}

// B fragment gathered when K runs along LDS rows (V tile stored [key][d])
static __device__ __forceinline__ v16bf ld_b_t(const bf16* base, int col, int kb,
                                               int stride) {
  union { v16bf v; bf16 h[16]; } r;
#pragma unroll
  for (int i = 0; i < 16; ++i) r.h[i] = base[(size_t)(kb + i) * stride + col];
  return r.v;
}

// ---------------------------------------------------------------------------
// Tiled GEMM: C[M,N] = A[M,K](bf16) * B[N,K](f32 weights)^T
// A tiles staged by the Tensor Data Mover with LDS ping-pong: tile t+1 DMA
// overlaps the 16-WMMA compute on tile t. B tiles staged by VALU with f32->bf16
// conversion. Block 256 threads = 8 waves; block tile 256x128; wave tile 64x64.
// MODE 0: C = acc        MODE 1: C = acc + resid        MODE 2: C += rowscale*acc
// ---------------------------------------------------------------------------
#define BM 256
#define BN 128
#define BK 32
#define LSTR 34  // LDS row stride in bf16 elems = 16 data DWORDs + 1 pad DWORD

template <int MODE>
__global__ __launch_bounds__(256) void gemm_bf16(
    const bf16* __restrict__ A, const float* __restrict__ B, float* __restrict__ C,
    int M, int N, int K, const float* __restrict__ resid,
    const float* __restrict__ rowscale, int sstride) {
  __shared__ bf16 As[2][BM * LSTR];
  __shared__ bf16 Bs[BN * LSTR];
  const int tid = threadIdx.x;
  const int lane = tid & 31;
  const int wave = tid >> 5;
  const int m0 = blockIdx.y * BM;
  const int n0 = blockIdx.x * BN;
  const int wm = (wave & 3) * 64;   // 4 waves along M, 64 rows each
  const int wn = (wave >> 2) * 64;  // 2 waves along N, 64 cols each
  const unsigned as_lds[2] = {(unsigned)(size_t)(void*)As[0],
                              (unsigned)(size_t)(void*)As[1]};

  v8f acc[4][4];
#pragma unroll
  for (int i = 0; i < 4; ++i)
#pragma unroll
    for (int j = 0; j < 4; ++j)
      acc[i][j] = v8f{0.f, 0.f, 0.f, 0.f, 0.f, 0.f, 0.f, 0.f};

  const int sr = tid >> 1;        // B staging row 0..127
  const int cb = (tid & 1) * 16;  // B staging col base 0/16

  // prologue: DMA first A tile (256x32 bf16, pad 1 DWORD / 16 DWORDs -> LSTR)
  if (wave == 0)
    tdm_load_2d_bf16(as_lds[0], A + (size_t)m0 * K, BK, BM,
                     (unsigned long long)K, (unsigned)K, (unsigned)(M - m0),
                     /*16DW*/ 3u, /*1DW*/ 0u);

  int cur = 0;
  for (int kb = 0; kb < K; kb += BK) {
    {  // stage B tile: f32 weights -> bf16 in LDS (halves LDS traffic)
      const float4* gp = (const float4*)(B + (size_t)(n0 + sr) * K + kb + cb);
      float4 f0 = gp[0], f1 = gp[1], f2 = gp[2], f3 = gp[3];
      bf16* lp = Bs + sr * LSTR + cb;
      lp[0] = (bf16)f0.x;  lp[1] = (bf16)f0.y;  lp[2] = (bf16)f0.z;  lp[3] = (bf16)f0.w;
      lp[4] = (bf16)f1.x;  lp[5] = (bf16)f1.y;  lp[6] = (bf16)f1.z;  lp[7] = (bf16)f1.w;
      lp[8] = (bf16)f2.x;  lp[9] = (bf16)f2.y;  lp[10] = (bf16)f2.z; lp[11] = (bf16)f2.w;
      lp[12] = (bf16)f3.x; lp[13] = (bf16)f3.y; lp[14] = (bf16)f3.z; lp[15] = (bf16)f3.w;
    }
    if (kb + BK < K)  // global_prefetch_b8 for the next weight tile
      __builtin_prefetch(B + (size_t)(n0 + sr) * K + kb + BK + cb, 0, 1);
    if (wave == 0) __builtin_amdgcn_s_wait_tensorcnt(0);  // A tile `kb` landed
    __syncthreads();

    // kick off DMA of the NEXT A tile into the other buffer; overlaps compute
    if (wave == 0 && kb + BK < K)
      tdm_load_2d_bf16(as_lds[cur ^ 1], A + (size_t)m0 * K + kb + BK, BK, BM,
                       (unsigned long long)K, (unsigned)(K - kb - BK),
                       (unsigned)(M - m0), 3u, 0u);

    const bf16* Ab = As[cur];
    v16bf af[4], bfr[4];
#pragma unroll
    for (int i = 0; i < 4; ++i) af[i] = ld_a(Ab, wm + i * 16, lane, LSTR);
#pragma unroll
    for (int j = 0; j < 4; ++j) bfr[j] = ld_b(Bs, wn + j * 16, lane, LSTR);
#pragma unroll
    for (int i = 0; i < 4; ++i)
#pragma unroll
      for (int j = 0; j < 4; ++j) acc[i][j] = wmma_bf16(af[i], bfr[j], acc[i][j]);
    cur ^= 1;
    __syncthreads();
  }

  // C/D layout: lane 0-15 -> M=r, N=lane; lane 16-31 -> M=8+r, N=lane-16
  const int half = lane >> 4;
  const int cl = lane & 15;
#pragma unroll
  for (int i = 0; i < 4; ++i)
#pragma unroll
    for (int j = 0; j < 4; ++j)
#pragma unroll
      for (int rr = 0; rr < 8; ++rr) {
        int row = m0 + wm + i * 16 + half * 8 + rr;
        int col = n0 + wn + j * 16 + cl;
        size_t idx = (size_t)row * N + col;
        float v = acc[i][j][rr];
        if (MODE == 0) C[idx] = v;
        else if (MODE == 1) C[idx] = v + resid[idx];
        else C[idx] += rowscale[(size_t)row * sstride] * v;
      }
}

// ---------------------------------------------------------------------------
// Flash attention (causal, GQA). Block = 4 waves; each wave owns 16 query rows.
// Q frags resident in VGPRs; K/V 32-key tiles DMA'd to LDS by the TDM
// (64 data DWORDs + 4 pad DWORDs per row -> KSTR); online softmax in regs.
// ---------------------------------------------------------------------------
#define KSTR 136
#define PSTR 34

__global__ __launch_bounds__(128) void flash_attn(
    const bf16* __restrict__ Q, const bf16* __restrict__ Kk,
    const bf16* __restrict__ Vv, bf16* __restrict__ O, int S, int H, int KVH) {
  __shared__ bf16 Ks[32 * KSTR];
  __shared__ bf16 Vs[32 * KSTR];
  __shared__ bf16 Ps[4 * 16 * PSTR];
  const int tid = threadIdx.x, lane = tid & 31, wave = tid >> 5;
  const int h = blockIdx.y;
  const int kvh = h / (H / KVH);
  const int qblk = blockIdx.x * 64;
  const int qrow0 = qblk + wave * 16;
  const size_t qstr = (size_t)H * 128;
  const size_t kstr = (size_t)KVH * 128;
  const int half = lane >> 4;
  const int cl = lane & 15;
  const unsigned ks_lds = (unsigned)(size_t)(void*)Ks;
  const unsigned vs_lds = (unsigned)(size_t)(void*)Vs;

  // Q fragments: 4 K-slices covering head_dim 128, loaded once from global
  v16bf qf[4];
  {
    const bf16* rowp = Q + (size_t)(qrow0 + cl) * qstr + (size_t)h * 128;
    const int kp = half * 8;
#pragma unroll
    for (int dk = 0; dk < 4; ++dk) {
      const unsigned* up = (const unsigned*)(rowp + dk * 32);
      union { v16bf v; unsigned u[8]; } rr;
#pragma unroll
      for (int j = 0; j < 8; ++j) rr.u[j] = up[kp + j];
      qf[dk] = rr.v;
    }
  }

  v8f oacc[8];
#pragma unroll
  for (int dj = 0; dj < 8; ++dj)
    oacc[dj] = v8f{0.f, 0.f, 0.f, 0.f, 0.f, 0.f, 0.f, 0.f};
  float mrow[8], lrow[8];
#pragma unroll
  for (int rr = 0; rr < 8; ++rr) { mrow[rr] = -1e30f; lrow[rr] = 0.f; }
  const float scale = 0.088388347648318447f;  // 1/sqrt(128)

  const int ktmax = (qblk + 63) / 32;  // causal bound (tile granularity)
  for (int kt = 0; kt <= ktmax; ++kt) {
    const int kbase = kt * 32;
    if (wave == 0) {  // DMA K and V tiles (32 keys x 128) into padded LDS rows
      const bf16* kg = Kk + (size_t)kbase * kstr + (size_t)kvh * 128;
      const bf16* vg = Vv + (size_t)kbase * kstr + (size_t)kvh * 128;
      tdm_load_2d_bf16(ks_lds, kg, 128, 32, (unsigned long long)kstr,
                       128u, (unsigned)(S - kbase), /*64DW*/ 5u, /*4DW*/ 3u);
      tdm_load_2d_bf16(vs_lds, vg, 128, 32, (unsigned long long)kstr,
                       128u, (unsigned)(S - kbase), /*64DW*/ 5u, /*4DW*/ 3u);
      __builtin_amdgcn_s_wait_tensorcnt(0);
    }
    __syncthreads();

    // S = Q Kt^T : 16x32 scores via 8 WMMAs (contraction over head_dim)
    v8f sa[2];
#pragma unroll
    for (int j = 0; j < 2; ++j) sa[j] = v8f{0.f, 0.f, 0.f, 0.f, 0.f, 0.f, 0.f, 0.f};
#pragma unroll
    for (int dk = 0; dk < 4; ++dk)
#pragma unroll
      for (int j = 0; j < 2; ++j) {
        v16bf kb_ = ld_b(Ks + dk * 32, j * 16, lane, KSTR);
        sa[j] = wmma_bf16(qf[dk], kb_, sa[j]);
      }

    // causal mask + online softmax (row stats live in 16-lane column groups)
    float pv[2][8];
#pragma unroll
    for (int j = 0; j < 2; ++j)
#pragma unroll
      for (int rr = 0; rr < 8; ++rr) {
        int krow = kbase + j * 16 + cl;
        int qrow = qrow0 + half * 8 + rr;
        float sv = sa[j][rr] * scale;
        pv[j][rr] = (krow <= qrow) ? sv : -1e30f;
      }
#pragma unroll
    for (int rr = 0; rr < 8; ++rr) {
      float rm = fmaxf(pv[0][rr], pv[1][rr]);
#pragma unroll
      for (int off = 1; off < 16; off <<= 1) rm = fmaxf(rm, __shfl_xor(rm, off, 32));
      float mn = fmaxf(mrow[rr], rm);
      float sc = __expf(mrow[rr] - mn);
      mrow[rr] = mn;
      float psum = 0.f;
#pragma unroll
      for (int j = 0; j < 2; ++j) {
        float p = (pv[j][rr] > -1e29f) ? __expf(pv[j][rr] - mn) : 0.f;
        pv[j][rr] = p;
        psum += p;
      }
#pragma unroll
      for (int off = 1; off < 16; off <<= 1) psum += __shfl_xor(psum, off, 32);
      lrow[rr] = lrow[rr] * sc + psum;
#pragma unroll
      for (int dj = 0; dj < 8; ++dj) oacc[dj][rr] *= sc;
    }

    // restage P (C-layout -> A-layout) through per-wave LDS slab
    bf16* pw = Ps + wave * 16 * PSTR;
#pragma unroll
    for (int j = 0; j < 2; ++j)
#pragma unroll
      for (int rr = 0; rr < 8; ++rr)
        pw[(half * 8 + rr) * PSTR + j * 16 + cl] = (bf16)pv[j][rr];
    __syncthreads();

    // O += P V : 8 WMMAs (contraction over 32 keys)
    v16bf pa = ld_a(pw, 0, lane, PSTR);
    const int kb2 = (lane < 16) ? 0 : 16;
#pragma unroll
    for (int dj = 0; dj < 8; ++dj) {
      v16bf vb = ld_b_t(Vs, dj * 16 + cl, kb2, KSTR);
      oacc[dj] = wmma_bf16(pa, vb, oacc[dj]);
    }
    __syncthreads();
  }

#pragma unroll
  for (int dj = 0; dj < 8; ++dj)
#pragma unroll
    for (int rr = 0; rr < 8; ++rr) {
      int qrow = qrow0 + half * 8 + rr;
      int d = dj * 16 + cl;
      O[(size_t)qrow * qstr + (size_t)h * 128 + d] = (bf16)(oacc[dj][rr] / lrow[rr]);
    }
}

// ---------------------------------------------------------------------------
// Elementwise / reduction helpers
// ---------------------------------------------------------------------------
__global__ __launch_bounds__(256) void rmsnorm_bf16(const float* __restrict__ X,
                                                    const float* __restrict__ W,
                                                    bf16* __restrict__ Y, int Hd) {
  int row = blockIdx.x;
  const float* xr = X + (size_t)row * Hd;
  float ss = 0.f;
  for (int i = threadIdx.x; i < Hd; i += blockDim.x) { float v = xr[i]; ss += v * v; }
#pragma unroll
  for (int off = 16; off > 0; off >>= 1) ss += __shfl_xor(ss, off, 32);
  __shared__ float red[8];
  if ((threadIdx.x & 31) == 0) red[threadIdx.x >> 5] = ss;
  __syncthreads();
  float tot = 0.f;
#pragma unroll
  for (int i = 0; i < 8; ++i) tot += red[i];
  float rs = rsqrtf(tot / (float)Hd + 1e-5f);
  for (int i = threadIdx.x; i < Hd; i += blockDim.x)
    Y[(size_t)row * Hd + i] = (bf16)(xr[i] * rs * W[i]);
}

__global__ void rope_to_bf16(const float* __restrict__ X, bf16* __restrict__ Y,
                             int S, int nH) {
  int gid = blockIdx.x * blockDim.x + threadIdx.x;
  int total = S * nH * 128;
  if (gid >= total) return;
  int d = gid & 127;
  int s = (gid >> 7) / nH;
  int base = gid - d;
  float x = X[gid];
  float other = (d < 64) ? -X[base + d + 64] : X[base + d - 64];
  // inv_freq = theta^(-2i/128), i = d mod 64; theta = 10000
  float inv = __expf(-(float)(2 * (d & 63)) * (1.0f / 128.0f) * 9.210340371976184f);
  float ang = (float)s * inv;
  float sn, c;
  __sincosf(ang, &sn, &c);
  Y[gid] = (bf16)(x * c + other * sn);
}

__global__ void cvt_bf16(const float* __restrict__ X, bf16* __restrict__ Y, int n) {
  int gid = blockIdx.x * blockDim.x + threadIdx.x;
  if (gid < n) Y[gid] = (bf16)X[gid];
}

__global__ void swiglu_bf16(const float* __restrict__ G, const float* __restrict__ U,
                            bf16* __restrict__ Y, int n) {
  int gid = blockIdx.x * blockDim.x + threadIdx.x;
  if (gid >= n) return;
  float g = G[gid];
  Y[gid] = (bf16)((g / (1.f + __expf(-g))) * U[gid]);
}

// softmax over E gate logits, top-2, normalized weights -> dense [T,E] mask
__global__ void gate_top2(const bf16* __restrict__ X, const float* __restrict__ GW,
                          float* __restrict__ TW, int Hd, int E) {
  int t = blockIdx.x;
  int lane = threadIdx.x & 31;
  int e = threadIdx.x >> 5;  // one wave per expert
  const bf16* xr = X + (size_t)t * Hd;
  const float* wr = GW + (size_t)e * Hd;
  float s = 0.f;
  for (int i = lane; i < Hd; i += 32) s += (float)xr[i] * wr[i];
#pragma unroll
  for (int off = 16; off > 0; off >>= 1) s += __shfl_xor(s, off, 32);
  __shared__ float lg[16];
  if (lane == 0) lg[e] = s;
  __syncthreads();
  if (threadIdx.x == 0) {
    float mx = lg[0];
    for (int i = 1; i < E; ++i) mx = fmaxf(mx, lg[i]);
    float p[16], sum = 0.f;
    for (int i = 0; i < E; ++i) { p[i] = __expf(lg[i] - mx); sum += p[i]; }
    for (int i = 0; i < E; ++i) p[i] /= sum;
    int i0 = 0;
    for (int i = 1; i < E; ++i) if (p[i] > p[i0]) i0 = i;
    int i1 = (i0 == 0) ? 1 : 0;
    for (int i = 0; i < E; ++i) if (i != i0 && p[i] > p[i1]) i1 = i;
    float norm = p[i0] + p[i1];
    for (int i = 0; i < E; ++i)
      TW[(size_t)t * E + i] =
          (i == i0) ? p[i0] / norm : (i == i1) ? p[i1] / norm : 0.f;
  }
}

// ---------------------------------------------------------------------------
extern "C" void kernel_launch(void* const* d_in, const int* in_sizes, int n_in,
                              void* d_out, int out_size, void* d_ws, size_t ws_size,
                              hipStream_t stream) {
  const float* hidden = (const float*)d_in[0];
  const float* ln1 = (const float*)d_in[1];
  const float* ln2 = (const float*)d_in[2];
  const float* q_w = (const float*)d_in[3];
  const float* k_w = (const float*)d_in[4];
  const float* v_w = (const float*)d_in[5];
  const float* o_w = (const float*)d_in[6];
  const float* gate_w = (const float*)d_in[7];
  const float* w1 = (const float*)d_in[8];  // dict order: w1, w3, w2
  const float* w3 = (const float*)d_in[9];
  const float* w2 = (const float*)d_in[10];

  const int Hd = in_sizes[1];
  const int S = in_sizes[0] / Hd;
  const int D = 128;
  const int H = Hd / D;
  const int KVd = in_sizes[4] / Hd;  // KVH * D
  const int KVH = KVd / D;
  const int E = in_sizes[7] / Hd;
  const int F = (int)((long long)in_sizes[8] / ((long long)E * Hd));

  // workspace carve (256B aligned)
  char* wp = (char*)d_ws;
  auto carve = [&](size_t bytes) -> void* {
    void* r = (void*)wp;
    wp += (bytes + 255) & ~(size_t)255;
    return r;
  };
  bf16* xn1 = (bf16*)carve((size_t)S * Hd * 2);
  float* qf = (float*)carve((size_t)S * Hd * 4);
  float* kf = (float*)carve((size_t)S * KVd * 4);
  float* vf = (float*)carve((size_t)S * KVd * 4);
  bf16* qb = (bf16*)carve((size_t)S * Hd * 2);
  bf16* kb = (bf16*)carve((size_t)S * KVd * 2);
  bf16* vb = (bf16*)carve((size_t)S * KVd * 2);
  bf16* attnb = (bf16*)carve((size_t)S * Hd * 2);
  float* hbuf = (float*)carve((size_t)S * Hd * 4);
  bf16* xn2 = (bf16*)carve((size_t)S * Hd * 2);
  float* topw = (float*)carve((size_t)S * E * 4);
  float* t1 = (float*)carve((size_t)S * F * 4);
  float* t3 = (float*)carve((size_t)S * F * 4);
  bf16* actb = (bf16*)carve((size_t)S * F * 2);

  // --- attention block ---
  rmsnorm_bf16<<<S, 256, 0, stream>>>(hidden, ln1, xn1, Hd);
  gemm_bf16<0><<<dim3(Hd / BN, S / BM), 256, 0, stream>>>(xn1, q_w, qf, S, Hd, Hd,
                                                          nullptr, nullptr, 0);
  gemm_bf16<0><<<dim3(KVd / BN, S / BM), 256, 0, stream>>>(xn1, k_w, kf, S, KVd, Hd,
                                                           nullptr, nullptr, 0);
  gemm_bf16<0><<<dim3(KVd / BN, S / BM), 256, 0, stream>>>(xn1, v_w, vf, S, KVd, Hd,
                                                           nullptr, nullptr, 0);
  {
    int nq = S * Hd, nk = S * KVd;
    rope_to_bf16<<<(nq + 255) / 256, 256, 0, stream>>>(qf, qb, S, H);
    rope_to_bf16<<<(nk + 255) / 256, 256, 0, stream>>>(kf, kb, S, KVH);
    cvt_bf16<<<(nk + 255) / 256, 256, 0, stream>>>(vf, vb, nk);
  }
  flash_attn<<<dim3(S / 64, H), 128, 0, stream>>>(qb, kb, vb, attnb, S, H, KVH);
  gemm_bf16<1><<<dim3(Hd / BN, S / BM), 256, 0, stream>>>(attnb, o_w, hbuf, S, Hd,
                                                          Hd, hidden, nullptr, 0);

  // --- MoE block ---
  rmsnorm_bf16<<<S, 256, 0, stream>>>(hbuf, ln2, xn2, Hd);
  gate_top2<<<S, 32 * E, 0, stream>>>(xn2, gate_w, topw, Hd, E);
  hipMemcpyAsync(d_out, hbuf, (size_t)S * Hd * sizeof(float),
                 hipMemcpyDeviceToDevice, stream);  // out = h (residual)
  for (int e = 0; e < E; ++e) {
    const float* w1e = w1 + (size_t)e * F * Hd;
    const float* w3e = w3 + (size_t)e * F * Hd;
    const float* w2e = w2 + (size_t)e * Hd * F;
    gemm_bf16<0><<<dim3(F / BN, S / BM), 256, 0, stream>>>(xn2, w1e, t1, S, F, Hd,
                                                           nullptr, nullptr, 0);
    gemm_bf16<0><<<dim3(F / BN, S / BM), 256, 0, stream>>>(xn2, w3e, t3, S, F, Hd,
                                                           nullptr, nullptr, 0);
    int nf = S * F;
    swiglu_bf16<<<(nf + 255) / 256, 256, 0, stream>>>(t1, t3, actb, nf);
    // out += topw[:,e] * (act @ w2[e]^T)
    gemm_bf16<2><<<dim3(Hd / BN, S / BM), 256, 0, stream>>>(
        actb, w2e, (float*)d_out, S, Hd, F, nullptr, topw + e, E);
  }
}
